// GraphAttentionEmbedding_52922587021370
// MI455X (gfx1250) — matrix operand order repaired and checked by reading, not compiled, not run on hardware
//
#include <hip/hip_runtime.h>
#include <hip/hip_bf16.h>

#define N_NODES 50000
#define N_EDGES 500000
#define DIM     128      // D == H*C
#define HC      128
#define TDIM    32
#define MDIM    64
#define KEDGE   96       // TD + MD

typedef __attribute__((ext_vector_type(16))) __bf16 v16bf;
typedef __attribute__((ext_vector_type(8)))  float  v8f;

// ---------------------------------------------------------------------------
// Weight packing: bf16, pre-swizzled into WMMA B-fragment layout.
// B-frag (16-bit B 32x16): lane l holds column n = l%16, K = kstep*32 + (l/16)*16 + i.
// Node pack: ntile in [0,32) over [Wq|Wk|Wv|Wskip] (8 ntiles each), 4 ksteps (K=128).
// Edge pack: ntile in [0,8) over We (128 cols), 3 ksteps (K=96).
// ---------------------------------------------------------------------------
__global__ void pack_weights(const float* Wq, const float* Wk, const float* Wv,
                             const float* Wskip, const float* We,
                             __bf16* wpack, __bf16* wepack) {
  int idx = blockIdx.x * 256 + threadIdx.x;
  if (idx < 32 * 4 * 32 * 16) {
    int ntile = idx >> 11;
    int r     = idx & 2047;
    int kstep = r >> 9;
    int r2    = r & 511;
    int lane  = r2 >> 4;
    int i     = r2 & 15;
    int ncol  = ((ntile & 7) << 4) + (lane & 15);
    int k     = kstep * 32 + ((lane >> 4) << 4) + i;
    int sel   = ntile >> 3;
    const float* W = (sel == 0) ? Wq : (sel == 1) ? Wk : (sel == 2) ? Wv : Wskip;
    wpack[idx] = (__bf16)W[k * HC + ncol];
  } else {
    int j = idx - 32 * 4 * 32 * 16;
    if (j >= 8 * 3 * 32 * 16) return;
    int ntile = j / 1536;
    int r     = j - ntile * 1536;
    int kstep = r >> 9;
    int r2    = r & 511;
    int lane  = r2 >> 4;
    int i     = r2 & 15;
    int ncol  = ntile * 16 + (lane & 15);
    int k     = kstep * 32 + ((lane >> 4) << 4) + i;
    wepack[j] = (__bf16)We[k * HC + ncol];
  }
}

__global__ void init_buf(unsigned int* nmax, float* denom, int n) {
  int i = blockIdx.x * 256 + threadIdx.x;
  if (i < n) { nmax[i] = 0u; denom[i] = 0.0f; }
}

// ---------------------------------------------------------------------------
// Node projections: q|k|v -> workspace, skip -> d_out.  16-row tile per block.
// ---------------------------------------------------------------------------
__global__ void __launch_bounds__(256)
node_proj(const float* __restrict__ x, const __bf16* __restrict__ wpack,
          const float* __restrict__ bq, const float* __restrict__ bk,
          const float* __restrict__ bv, const float* __restrict__ bskip,
          float* __restrict__ qb, float* __restrict__ kb,
          float* __restrict__ vb, float* __restrict__ outp) {
  __shared__ __bf16 sx[16][DIM];
  int tid = threadIdx.x;
  size_t rowBase = (size_t)blockIdx.x * 16;

  for (int idx = tid; idx < 16 * DIM; idx += 256) {
    int r = idx >> 7, c = idx & 127;
    sx[r][c] = (__bf16)x[(rowBase + r) * DIM + c];
  }
  __syncthreads();

  int wave = tid >> 5, lane = tid & 31;
  int m = lane & 15;

  // A-frag (16-bit A 16x32): lane l holds row m=l%16, K = kb..kb+15,
  // kb = kstep*32 + (l/16)*16.  Preload all 4 ksteps.
  v16bf afr[4];
#pragma unroll
  for (int s = 0; s < 4; ++s) {
    int kbase = s * 32 + ((lane >> 4) << 4);
#pragma unroll
    for (int i = 0; i < 8; ++i) {
      afr[s][i]     = sx[m][kbase + i];
      afr[s][i + 8] = sx[m][kbase + 8 + i];
    }
  }

#pragma unroll
  for (int t = 0; t < 4; ++t) {
    int ntile = wave + t * 8;
    v8f c = {};
#pragma unroll
    for (int s = 0; s < 4; ++s) {
      v16bf bfr = *(const v16bf*)(wpack + (((size_t)(ntile * 4 + s)) * 32 + lane) * 16);
      c = __builtin_amdgcn_wmma_f32_16x16x32_bf16(false, afr[s], false, bfr,
                                                  (short)0, c, false, false);
    }
    int sel = ntile >> 3;
    int col = ((ntile & 7) << 4) + (lane & 15);
    const float* bias = (sel == 0) ? bq : (sel == 1) ? bk : (sel == 2) ? bv : bskip;
    float* dst = (sel == 0) ? qb : (sel == 1) ? kb : (sel == 2) ? vb : outp;
    float b = bias[col];
#pragma unroll
    for (int j = 0; j < 8; ++j) {
      int row = ((lane >> 4) << 3) + j;
      dst[(rowBase + row) * HC + col] = c[j] + b;
    }
  }
}

// ---------------------------------------------------------------------------
// Shared edge helpers
// ---------------------------------------------------------------------------
__device__ inline void stage_edge_attr(int tid, size_t eBase,
                                       const int* __restrict__ etime,
                                       const float* __restrict__ msg,
                                       const float* __restrict__ Wt,
                                       const float* __restrict__ bt,
                                       __bf16 (*sattr)[KEDGE]) {
  for (int idx = tid; idx < 16 * TDIM; idx += 256) {
    int el = idx >> 5, j = idx & 31;
    float t = (float)etime[eBase + el];
    sattr[el][j] = (__bf16)__cosf(fmaf(t, Wt[j], bt[j]));
  }
  for (int idx = tid; idx < 16 * MDIM; idx += 256) {
    int el = idx >> 6, j = idx & 63;
    sattr[el][TDIM + j] = (__bf16)msg[(eBase + el) * MDIM + j];
  }
}

__device__ inline v8f edge_wmma(const __bf16 (*sattr)[KEDGE],
                                const __bf16* __restrict__ wepack,
                                int ntile, int lane) {
  v8f c = {};
  int m = lane & 15;
#pragma unroll
  for (int s = 0; s < 3; ++s) {
    v16bf a;
    int kbase = s * 32 + ((lane >> 4) << 4);
#pragma unroll
    for (int i = 0; i < 8; ++i) {
      a[i]     = sattr[m][kbase + i];
      a[i + 8] = sattr[m][kbase + 8 + i];
    }
    v16bf b = *(const v16bf*)(wepack + (((size_t)(ntile * 3 + s)) * 32 + lane) * 16);
    c = __builtin_amdgcn_wmma_f32_16x16x32_bf16(false, a, false, b,
                                                (short)0, c, false, false);
  }
  return c;
}

// ---------------------------------------------------------------------------
// Pass 1 over edges: attention logits + segment max (ordered-uint atomicMax)
// ---------------------------------------------------------------------------
__global__ void __launch_bounds__(256)
edge_logits(const int* __restrict__ ei, const int* __restrict__ etime,
            const float* __restrict__ msg, const float* __restrict__ Wt,
            const float* __restrict__ bt, const __bf16* __restrict__ wepack,
            const float* __restrict__ qb, const float* __restrict__ kb,
            float* __restrict__ alphab, unsigned int* __restrict__ nmax) {
  __shared__ __bf16 sattr[16][KEDGE];
  __shared__ float sq[16][HC];
  __shared__ float sk[16][HC];
  __shared__ float salpha[16][2];
  __shared__ int ssrc[16], sdst[16];

  int tid = threadIdx.x;
  size_t eBase = (size_t)blockIdx.x * 16;

  if (tid < 16) { ssrc[tid] = ei[eBase + tid]; sdst[tid] = ei[N_EDGES + eBase + tid]; }
  if (tid < 32) salpha[tid >> 1][tid & 1] = 0.0f;
  stage_edge_attr(tid, eBase, etime, msg, Wt, bt, sattr);
  __syncthreads();

  for (int idx = tid; idx < 16 * HC; idx += 256) {
    int r = idx >> 7, c = idx & 127;
    sq[r][c] = qb[(size_t)sdst[r] * HC + c];
    sk[r][c] = kb[(size_t)ssrc[r] * HC + c];
  }
  __syncthreads();

  int wave = tid >> 5, lane = tid & 31;
  v8f c = edge_wmma(sattr, wepack, wave, lane);

  int head = wave >> 2;                  // cols [wave*16, wave*16+16) -> head
  int col  = wave * 16 + (lane & 15);
#pragma unroll
  for (int j = 0; j < 8; ++j) {
    int m = ((lane >> 4) << 3) + j;
    float part = sq[m][col] * (sk[m][col] + c[j]);
    part += __shfl_xor(part, 8, 16);
    part += __shfl_xor(part, 4, 16);
    part += __shfl_xor(part, 2, 16);
    part += __shfl_xor(part, 1, 16);
    if ((lane & 15) == 0) atomicAdd(&salpha[m][head], part);
  }
  __syncthreads();

  if (tid < 32) {
    int el = tid >> 1, h = tid & 1;
    float logit = salpha[el][h] * 0.125f;   // 1/sqrt(C), C=64
    alphab[(eBase + el) * 2 + h] = logit;
    unsigned int u = __float_as_uint(logit);
    unsigned int key = (u & 0x80000000u) ? ~u : (u | 0x80000000u);
    atomicMax(&nmax[(size_t)sdst[el] * 2 + h], key);
  }
}

// ---------------------------------------------------------------------------
// Pass 2: ex = exp(alpha - max[dst]); denom += ex
// ---------------------------------------------------------------------------
__global__ void edge_exp(const int* __restrict__ ei,
                         const unsigned int* __restrict__ nmax,
                         float* __restrict__ alphab, float* __restrict__ denom) {
  int idx = blockIdx.x * 256 + threadIdx.x;
  if (idx >= N_EDGES * 2) return;
  int e = idx >> 1, h = idx & 1;
  int d = ei[N_EDGES + e];
  unsigned int key = nmax[(size_t)d * 2 + h];
  float mx = 0.0f;
  if (key != 0u) {
    unsigned int u = (key & 0x80000000u) ? (key ^ 0x80000000u) : ~key;
    float f = __uint_as_float(u);
    mx = isfinite(f) ? f : 0.0f;
  }
  float ex = __expf(alphab[idx] - mx);
  alphab[idx] = ex;
  unsafeAtomicAdd(&denom[(size_t)d * 2 + h], ex);
}

// ---------------------------------------------------------------------------
// Pass 3: out[dst] += (v[src] + e) * ex/(denom+eps)   (e recomputed via WMMA)
// ---------------------------------------------------------------------------
__global__ void __launch_bounds__(256)
edge_aggregate(const int* __restrict__ ei, const int* __restrict__ etime,
               const float* __restrict__ msg, const float* __restrict__ Wt,
               const float* __restrict__ bt, const __bf16* __restrict__ wepack,
               const float* __restrict__ vb, const float* __restrict__ alphab,
               const float* __restrict__ denom, float* __restrict__ outp) {
  __shared__ __bf16 sattr[16][KEDGE];
  __shared__ float sv[16][HC];
  __shared__ float scoef[16][2];
  __shared__ int ssrc[16], sdst[16];

  int tid = threadIdx.x;
  size_t eBase = (size_t)blockIdx.x * 16;

  if (tid < 16) { ssrc[tid] = ei[eBase + tid]; sdst[tid] = ei[N_EDGES + eBase + tid]; }
  stage_edge_attr(tid, eBase, etime, msg, Wt, bt, sattr);
  __syncthreads();

  for (int idx = tid; idx < 16 * HC; idx += 256) {
    int r = idx >> 7, c = idx & 127;
    sv[r][c] = vb[(size_t)ssrc[r] * HC + c];
  }
  if (tid < 32) {
    int el = tid >> 1, h = tid & 1;
    float den = denom[(size_t)sdst[el] * 2 + h];
    scoef[el][h] = alphab[(eBase + el) * 2 + h] / (den + 1e-16f);
  }
  __syncthreads();

  int wave = tid >> 5, lane = tid & 31;
  v8f c = edge_wmma(sattr, wepack, wave, lane);

  int head = wave >> 2;
  int col  = wave * 16 + (lane & 15);
#pragma unroll
  for (int j = 0; j < 8; ++j) {
    int m = ((lane >> 4) << 3) + j;
    float val = (sv[m][col] + c[j]) * scoef[m][head];
    unsafeAtomicAdd(&outp[(size_t)sdst[m] * HC + col], val);
  }
}

// ---------------------------------------------------------------------------
extern "C" void kernel_launch(void* const* d_in, const int* in_sizes, int n_in,
                              void* d_out, int out_size, void* d_ws, size_t ws_size,
                              hipStream_t stream) {
  const float* x     = (const float*)d_in[0];
  const int*   ei    = (const int*)d_in[1];
  const int*   etime = (const int*)d_in[2];
  const float* msg   = (const float*)d_in[3];
  const float* Wt    = (const float*)d_in[4];
  const float* bt    = (const float*)d_in[5];
  const float* Wq    = (const float*)d_in[6];
  const float* bq    = (const float*)d_in[7];
  const float* Wk    = (const float*)d_in[8];
  const float* bk    = (const float*)d_in[9];
  const float* Wv    = (const float*)d_in[10];
  const float* bv    = (const float*)d_in[11];
  const float* We    = (const float*)d_in[12];
  const float* Wskip = (const float*)d_in[13];
  const float* bskip = (const float*)d_in[14];
  float* outp = (float*)d_out;

  // workspace layout (offsets 256B-aligned, denom only 128B-aligned)
  char* ws = (char*)d_ws;
  const size_t szNode = (size_t)N_NODES * HC * sizeof(float);     // 25.6 MB
  float*        qb     = (float*)(ws);
  float*        kb     = (float*)(ws + szNode);
  float*        vb     = (float*)(ws + 2 * szNode);
  __bf16*       wpack  = (__bf16*)(ws + 3 * szNode);              // 128 KB
  __bf16*       wepack = (__bf16*)(ws + 3 * szNode + 131072);     // 24 KB
  float*        alphab = (float*)(ws + 3 * szNode + 131072 + 24576);       // E*2 f32
  unsigned int* nmax   = (unsigned int*)(ws + 3 * szNode + 131072 + 24576
                                         + (size_t)N_EDGES * 2 * sizeof(float));
  float*        denom  = (float*)((char*)nmax + (size_t)N_NODES * 2 * sizeof(unsigned int));

  // 1. pack weights to bf16 WMMA fragment layout
  pack_weights<<<(32*4*32*16 + 8*3*32*16 + 255) / 256, 256, 0, stream>>>(
      Wq, Wk, Wv, Wskip, We, wpack, wepack);

  // 2. init segment max / denom
  init_buf<<<(N_NODES * 2 + 255) / 256, 256, 0, stream>>>(nmax, denom, N_NODES * 2);

  // 3. node projections (q,k,v -> ws; skip -> d_out)
  node_proj<<<N_NODES / 16, 256, 0, stream>>>(x, wpack, bq, bk, bv, bskip,
                                              qb, kb, vb, outp);

  // 4. edge logits + segment max
  edge_logits<<<N_EDGES / 16, 256, 0, stream>>>(ei, etime, msg, Wt, bt, wepack,
                                                qb, kb, alphab, nmax);

  // 5. exp + denom
  edge_exp<<<(N_EDGES * 2 + 255) / 256, 256, 0, stream>>>(ei, nmax, alphab, denom);

  // 6. weighted aggregation into d_out
  edge_aggregate<<<N_EDGES / 16, 256, 0, stream>>>(ei, etime, msg, Wt, bt, wepack,
                                                   vb, alphab, denom, outp);
}
